// DCM_979252544278
// MI455X (gfx1250) — compile-verified, matrix operand order checked by reading, not run on hardware
//
#include <hip/hip_runtime.h>
#include <hip/hip_bf16.h>
#include <math.h>

#define BB 64
#define CC 21
#define LL 8192
#define DD 512
#define KG 25
#define KP 15

#define PI_F      3.14159265358979323846f
#define TWO_PI_F  6.28318530717958647692f

typedef __attribute__((ext_vector_type(16))) _Float16 v16h;
typedef __attribute__((ext_vector_type(8)))  _Float16 v8h;
typedef __attribute__((ext_vector_type(2)))  _Float16 v2h;
typedef __attribute__((ext_vector_type(8)))  float    v8f;

__device__ __forceinline__ float gelu_exact(float x) {
    return 0.5f * x * (1.0f + erff(x * 0.70710678118654752f));
}

__device__ __forceinline__ float softplusf(float x) {
    return (x > 20.0f) ? x : log1pf(expf(x));
}

// Split two fp32 values into f16 hi + f16 lo pairs packed for b32 LDS stores.
__device__ __forceinline__ void split2(float x0, float x1, v2h* hi, v2h* lo) {
    _Float16 h0 = (_Float16)x0;
    _Float16 h1 = (_Float16)x1;
    (*hi)[0] = h0;
    (*hi)[1] = h1;
    (*lo)[0] = (_Float16)(x0 - (float)h0);
    (*lo)[1] = (_Float16)(x1 - (float)h1);
}

// Fragment load: ISA 7.12.2 16-bit layout, element e -> k = (e/8)*16 + half*8 + e%8.
// With tiles stored so that the 32 K-values of one row/col are contiguous, each
// lane's fragment is two aligned 16-byte chunks -> 2x ds_load_b128.
__device__ __forceinline__ v16h load_frag16(const _Float16* p, int fh) {
    v8h lo = *(const v8h*)(p + (fh << 3));
    v8h hi = *(const v8h*)(p + 16 + (fh << 3));
    return __builtin_shufflevector(lo, hi, 0, 1, 2, 3, 4, 5, 6, 7,
                                   8, 9, 10, 11, 12, 13, 14, 15);
}

// ---------------------------------------------------------------------------
// GEMM + bias + exact GELU.  C[M,N] = gelu(A[M,K] @ B[K,N] + bias)
// Split-f16 precision: v = hi + lo (both f16); acc += hi*hi + hi*lo + lo*hi.
// 64x64 block tile, 8 waves; each wave owns two 16x16 WMMA accumulators.
// A tile row-major [row][k], B tile TRANSPOSED in LDS [col][k]; both padded to
// stride 40 halves (80 B: 16B-aligned, bank-rotating). Register double-buffer
// on the global tile loads; per-thread K-pairs so LDS stores are b32.
// ---------------------------------------------------------------------------
__global__ __launch_bounds__(256)
void gemm_gelu_kernel(const float* __restrict__ A, const float* __restrict__ Bw,
                      const float* __restrict__ bias, float* __restrict__ out,
                      int M, int N, int K) {
    constexpr int TM = 64, TN = 64, TK = 32, ST = 40;  // ST = padded K-stride
    __shared__ __align__(16) _Float16 AsH[TM * ST];
    __shared__ __align__(16) _Float16 AsL[TM * ST];
    __shared__ __align__(16) _Float16 BsH[TN * ST];    // [col][k]
    __shared__ __align__(16) _Float16 BsL[TN * ST];

    const int tid  = threadIdx.x;
    const int lane = tid & 31;           // wave32
    const int wave = tid >> 5;           // 8 waves
    const int m0   = blockIdx.x * TM;
    const int n0   = blockIdx.y * TN;

    const int sm = wave & 3;             // sub-tile row   0..3
    const int sn = (wave >> 2) * 2;      // sub-tile cols  sn, sn+1
    const int fr = lane & 15;            // fragment row/col index
    const int fh = lane >> 4;            // lane half

    // per-thread global load coordinates: 4 K-pairs each for A and B
    const int ar2 = tid >> 4;            // A row    (+16j)
    const int ak2 = (tid & 15) * 2;      // A k-pair base
    const int bc2 = tid & 63;            // B col
    const int bk2 = (tid >> 6) * 2;      // B k-pair base (+8j)

    v8f acc0 = {};
    v8f acc1 = {};

    float2 aReg[4];
    float  bReg[4][2];
    // prologue: fetch k0 = 0 tiles into registers
#pragma unroll
    for (int j = 0; j < 4; ++j) {
        aReg[j] = *(const float2*)&A[(size_t)(m0 + ar2 + 16 * j) * K + ak2];
        bReg[j][0] = Bw[(size_t)(bk2 + 8 * j) * N + (n0 + bc2)];
        bReg[j][1] = Bw[(size_t)(bk2 + 8 * j + 1) * N + (n0 + bc2)];
    }

    for (int k0 = 0; k0 < K; k0 += TK) {
        // commit registers -> LDS with f32 = hi + lo split (b32 stores)
#pragma unroll
        for (int j = 0; j < 4; ++j) {
            v2h hi, lo;
            split2(aReg[j].x, aReg[j].y, &hi, &lo);
            *(v2h*)&AsH[(ar2 + 16 * j) * ST + ak2] = hi;
            *(v2h*)&AsL[(ar2 + 16 * j) * ST + ak2] = lo;
            split2(bReg[j][0], bReg[j][1], &hi, &lo);
            *(v2h*)&BsH[bc2 * ST + bk2 + 8 * j] = hi;    // transposed [col][k]
            *(v2h*)&BsL[bc2 * ST + bk2 + 8 * j] = lo;
        }
        __syncthreads();

        // overlap: fetch next tiles while this one is consumed
        if (k0 + TK < K) {
            const int kn = k0 + TK;
#pragma unroll
            for (int j = 0; j < 4; ++j) {
                aReg[j] = *(const float2*)&A[(size_t)(m0 + ar2 + 16 * j) * K + (kn + ak2)];
                bReg[j][0] = Bw[(size_t)(kn + bk2 + 8 * j) * N + (n0 + bc2)];
                bReg[j][1] = Bw[(size_t)(kn + bk2 + 8 * j + 1) * N + (n0 + bc2)];
            }
            __builtin_prefetch(&A[(size_t)(m0 + ar2) * K + (kn + TK + ak2)], 0, 3);
            __builtin_prefetch(&Bw[(size_t)(kn + TK + bk2) * N + n0 + bc2], 0, 3);
        }

        // fragments: 2x ds_load_b128 each
        const _Float16* aBH  = &AsH[(sm * 16 + fr) * ST];
        const _Float16* aBL  = &AsL[(sm * 16 + fr) * ST];
        const _Float16* b0BH = &BsH[(sn * 16 + fr) * ST];
        const _Float16* b0BL = &BsL[(sn * 16 + fr) * ST];
        const _Float16* b1BH = b0BH + 16 * ST;
        const _Float16* b1BL = b0BL + 16 * ST;

        v16h aH  = load_frag16(aBH, fh);
        v16h aL  = load_frag16(aBL, fh);
        v16h b0H = load_frag16(b0BH, fh);
        v16h b0L = load_frag16(b0BL, fh);
        v16h b1H = load_frag16(b1BH, fh);
        v16h b1L = load_frag16(b1BL, fh);

        acc0 = __builtin_amdgcn_wmma_f32_16x16x32_f16(false, aH, false, b0H, (short)0, acc0, false, false);
        acc0 = __builtin_amdgcn_wmma_f32_16x16x32_f16(false, aH, false, b0L, (short)0, acc0, false, false);
        acc0 = __builtin_amdgcn_wmma_f32_16x16x32_f16(false, aL, false, b0H, (short)0, acc0, false, false);
        acc1 = __builtin_amdgcn_wmma_f32_16x16x32_f16(false, aH, false, b1H, (short)0, acc1, false, false);
        acc1 = __builtin_amdgcn_wmma_f32_16x16x32_f16(false, aH, false, b1L, (short)0, acc1, false, false);
        acc1 = __builtin_amdgcn_wmma_f32_16x16x32_f16(false, aL, false, b1H, (short)0, acc1, false, false);
        __syncthreads();
    }

    // Epilogue: C/D layout — VGPR r holds row (r + 8*half), col = lane&15
    const int orow = m0 + sm * 16 + fh * 8;
    const int col0 = n0 + sn * 16 + fr;
    const int col1 = col0 + 16;
    const float bias0 = bias[col0];
    const float bias1 = bias[col1];
#pragma unroll
    for (int r = 0; r < 8; ++r) {
        int row = orow + r;
        out[(size_t)row * N + col0] = gelu_exact(acc0[r] + bias0);
        out[(size_t)row * N + col1] = gelu_exact(acc1[r] + bias1);
    }
}

// ---------------------------------------------------------------------------
// Gaussian decomposition: one workgroup per (b,c) row.
// seasonal = x - trend;  for b==0 also fuse A_t = alpha*beta1*softplus(beta2*clamp(trend))
// ---------------------------------------------------------------------------
__global__ __launch_bounds__(256)
void decomp_kernel(const float* __restrict__ x, const float* __restrict__ log_sigma,
                   const float* __restrict__ alpha_log, const float* __restrict__ beta1_log,
                   const float* __restrict__ beta2_log,
                   float* __restrict__ seasonal, float* __restrict__ At) {
    __shared__ float row[LL];
    const int r = blockIdx.x;
    const int b = r / CC;
    const int c = r % CC;
    const int tid = threadIdx.x;
    const float* xr = x + (size_t)r * LL;

    for (int t = tid; t < LL; t += 256) row[t] = xr[t];

    // per-channel gaussian taps
    float g[KG];
    float sigma  = expf(log_sigma[c]) + 1e-6f;
    float inv2s2 = 1.0f / (2.0f * sigma * sigma);
    float gsum = 0.0f;
#pragma unroll
    for (int j = 0; j < KG; ++j) {
        float d = (float)(j - KG / 2);
        g[j] = expf(-d * d * inv2s2);
        gsum += g[j];
    }
    float gn = 1.0f / (gsum + 1e-12f);
#pragma unroll
    for (int j = 0; j < KG; ++j) g[j] *= gn;

    const float beta1 = softplusf(beta1_log[0]) + 1e-6f;
    const float beta2 = softplusf(beta2_log[0]) + 1e-6f;
    const float alpha = softplusf(alpha_log[c]) + 1e-6f;

    __syncthreads();
    for (int t = tid; t < LL; t += 256) {
        float acc = 0.0f;
#pragma unroll
        for (int j = 0; j < KG; ++j) {
            int idx = t + j - KG / 2;
            idx = (idx < 0) ? -idx : idx;
            idx = (idx >= LL) ? (2 * LL - 2 - idx) : idx;
            acc += g[j] * row[idx];
        }
        seasonal[(size_t)r * LL + t] = row[t] - acc;
        if (b == 0) {
            float T = fminf(fmaxf(acc, -10.0f), 10.0f);
            At[(size_t)c * LL + t] = alpha * beta1 * softplusf(beta2 * T);
        }
    }
}

// ---------------------------------------------------------------------------
// Radix-2 Stockham autosort FFT over 8192 complex points in LDS (13 stages).
// sign = -1 forward, +1 inverse (1/N scale omitted: atan2 is scale-invariant).
// ---------------------------------------------------------------------------
__device__ float2* fft_stages(float2* src, float2* dst, float sign, int tid) {
    int l = LL / 2;
    int mshift = 0;
#pragma unroll 1
    for (int s = 0; s < 13; ++s) {
        __syncthreads();
        const int m = 1 << mshift;
        for (int t = tid; t < LL / 2; t += 256) {
            int p = t >> mshift;
            int q = t & (m - 1);
            float ang = sign * (PI_F * (float)p / (float)l);
            float sn, cs;
            sincosf(ang, &sn, &cs);
            float2 u = src[t];
            float2 v = src[t + LL / 2];
            float2 su, df, tw;
            su.x = u.x + v.x;  su.y = u.y + v.y;
            df.x = u.x - v.x;  df.y = u.y - v.y;
            tw.x = df.x * cs - df.y * sn;
            tw.y = df.x * sn + df.y * cs;
            int d0 = 2 * t - q;              // q + 2*m*p
            dst[d0]     = su;
            dst[d0 + m] = tw;
        }
        l >>= 1;
        ++mshift;
        float2* tmp = src; src = dst; dst = tmp;
    }
    __syncthreads();
    return src;   // buffer holding the result
}

__device__ __forceinline__ float unwrap_adj(const float* ph, int k) {
    float d  = ph[k + 1] - ph[k];
    float xx = d + PI_F;
    float fm = xx - TWO_PI_F * floorf(xx / TWO_PI_F);   // floor-mod like jnp.mod
    float dmod = fm - PI_F;
    if (dmod == -PI_F && d > 0.0f) dmod = PI_F;
    return dmod - d;
}

// ---------------------------------------------------------------------------
// Hilbert analytic signal -> phase -> unwrap -> zero-mean corrector -> I.
// One workgroup per (b,c) row; data overwritten in place (seasonal -> I).
// Dynamic LDS: two float2[8192] ping-pong buffers (128 KB).
// ---------------------------------------------------------------------------
__global__ __launch_bounds__(256)
void hilbert_kernel(float* __restrict__ data, const float* __restrict__ At,
                    const float* __restrict__ pcw, const float* __restrict__ pcs,
                    const float* __restrict__ phi0) {
    extern __shared__ float2 sm2[];
    float2* bufA = sm2;
    float2* bufB = sm2 + LL;
    __shared__ float tsum[256];

    const int r   = blockIdx.x;
    const int c   = r % CC;
    const int tid = threadIdx.x;
    float* grow = data + (size_t)r * LL;

    for (int t = tid; t < LL; t += 256) { bufA[t].x = grow[t]; bufA[t].y = 0.0f; }

    // forward FFT
    float2* F = fft_stages(bufA, bufB, -1.0f, tid);
    // Hilbert multiplier: h[0]=h[N/2]=1, h[1..N/2-1]=2, rest 0
    for (int t = tid; t < LL; t += 256) {
        float sc = (t == 0 || t == LL / 2) ? 1.0f : ((t < LL / 2) ? 2.0f : 0.0f);
        F[t].x *= sc;
        F[t].y *= sc;
    }
    float2* other = (F == bufA) ? bufB : bufA;
    float2* Z   = fft_stages(F, other, 1.0f, tid);      // inverse (unscaled)
    float2* scr = (Z == bufA) ? bufB : bufA;            // free 64 KB buffer
    float* ph = (float*)scr;                            // 32 KB: wrapped phase
    float* pu = ph + LL;                                // 32 KB: unwrapped phase

    for (int t = tid; t < LL; t += 256) ph[t] = atan2f(Z[t].y, Z[t].x);
    __syncthreads();

    // ---- phase unwrap: block scan over per-thread chunks of 32 ----
    const int CH = LL / 256;
    const int cb = tid * CH;
    float tot = 0.0f;
    for (int k = cb; k < cb + CH; ++k)
        if (k < LL - 1) tot += unwrap_adj(ph, k);
    tsum[tid] = tot;
    __syncthreads();
    if (tid == 0) {
        float run = 0.0f;
        for (int i = 0; i < 256; ++i) { float v = tsum[i]; tsum[i] = run; run += v; }
    }
    __syncthreads();
    float run = tsum[tid];
    for (int t = cb; t < cb + CH; ++t) {
        pu[t] = ph[t] + run;
        if (t < LL - 1) run += unwrap_adj(ph, t);
    }
    __syncthreads();

    // ---- zero-mean phase corrector + amplitude coupling ----
    float wc[KP];
    float wm = 0.0f;
#pragma unroll
    for (int j = 0; j < KP; ++j) { wc[j] = pcw[c * KP + j]; wm += wc[j]; }
    wm *= (1.0f / (float)KP);
#pragma unroll
    for (int j = 0; j < KP; ++j) wc[j] -= wm;
    const float ts = tanhf(pcs[0]);
    const float p0 = phi0[c];
    const float* Atc = At + (size_t)c * LL;

    for (int t = tid; t < LL; t += 256) {
        float delta = 0.0f;
#pragma unroll
        for (int j = 0; j < KP; ++j) {
            int idx = t + j - KP / 2;
            idx = (idx < 0) ? -idx : idx;
            idx = (idx >= LL) ? (2 * LL - 2 - idx) : idx;
            delta += wc[j] * pu[idx];
        }
        float phi = pu[t] + ts * delta + p0;
        grow[t] = Atc[t] * cosf(phi);
    }
}

// ---------------------------------------------------------------------------
extern "C" void kernel_launch(void* const* d_in, const int* in_sizes, int n_in,
                              void* d_out, int out_size, void* d_ws, size_t ws_size,
                              hipStream_t stream) {
    (void)in_sizes; (void)n_in; (void)out_size; (void)ws_size;

    const float* x_input     = (const float*)d_in[0];
    const float* x_w         = (const float*)d_in[1];
    const float* x_b         = (const float*)d_in[2];
    const float* i_w         = (const float*)d_in[3];
    const float* i_b         = (const float*)d_in[4];
    const float* log_sigma   = (const float*)d_in[5];
    const float* pc_weight   = (const float*)d_in[6];
    const float* pc_strength = (const float*)d_in[7];
    const float* alpha_log   = (const float*)d_in[8];
    const float* phi0        = (const float*)d_in[9];
    const float* beta1_log   = (const float*)d_in[10];
    const float* beta2_log   = (const float*)d_in[11];
    float* out = (float*)d_out;

    const int M = BB * CC;                       // 1344
    float* ws_I  = (float*)d_ws;                 // [B*C, L]   seasonal -> I (in place)
    float* ws_At = ws_I + (size_t)M * LL;        // [C, L]     amplitude from batch 0

    // 1) x_out = gelu(x @ x_w + x_b)
    gemm_gelu_kernel<<<dim3(M / 64, DD / 64), 256, 0, stream>>>(
        x_input, x_w, x_b, out, M, DD, LL);

    // 2) gaussian decomposition -> seasonal, A_t
    decomp_kernel<<<M, 256, 0, stream>>>(
        x_input, log_sigma, alpha_log, beta1_log, beta2_log, ws_I, ws_At);

    // 3) hilbert + unwrap + corrector + I  (in place on ws_I)
    hilbert_kernel<<<M, 256, 2 * LL * sizeof(float2), stream>>>(
        ws_I, ws_At, pc_weight, pc_strength, phi0);

    // 4) I_coupled = gelu(I @ i_w + i_b)
    gemm_gelu_kernel<<<dim3(M / 64, DD / 64), 256, 0, stream>>>(
        ws_I, i_w, i_b, out + (size_t)M * DD, M, DD, LL);
}
